// RBFLayer_35356170781238
// MI455X (gfx1250) — matrix operand order repaired and checked by reading, not compiled, not run on hardware
//
#include <hip/hip_runtime.h>
#include <math.h>

// ---------- types ----------
typedef __attribute__((ext_vector_type(16))) __bf16          v16bf;
typedef __attribute__((ext_vector_type(8)))  float           v8f;
typedef __attribute__((ext_vector_type(8)))  unsigned short  su8;   // 16B
typedef __attribute__((ext_vector_type(4)))  unsigned short  su4;   // 8B
typedef __attribute__((ext_vector_type(4)))  unsigned int    v4u;
typedef __attribute__((ext_vector_type(8)))  int             v8i;
typedef __attribute__((ext_vector_type(4)))  int             v4i;

// ---------- problem constants ----------
#define B_ROWS 8192
#define D_DIM  1024
#define U_COLS 2048

// ---------- tiling ----------
#define BM 128
#define BN 128
#define BK 32
#define PITCH 40                                 // ushorts per LDS row (32 + 8 pad)
#define A_HI  0
#define A_LO  (BM * PITCH)                       // 5120
#define BT_HI (2 * BM * PITCH)                   // 10240
#define BT_LO (2 * BM * PITCH + BN * PITCH)      // 15360
#define STAGE (2 * BM * PITCH + 2 * BN * PITCH)  // 20480 ushorts = 40960 B

#if defined(__has_builtin)
#  if __has_builtin(__builtin_amdgcn_tensor_load_to_lds)
#    define HAVE_TDM 1
#  endif
#endif
#ifndef HAVE_TDM
#  define HAVE_TDM 0
#endif

union Frag { v16bf v; su8 u[2]; };

// exact fp32 -> (hi,lo) bf16 split: x = hi + lo up to ~2^-16 relative
__device__ __forceinline__ void bsplit(float x, unsigned short &h, unsigned short &l) {
    unsigned xb = __float_as_uint(x);
    float hf = __uint_as_float(xb & 0xffff0000u);   // truncated hi
    h = (unsigned short)(xb >> 16);
    l = (unsigned short)(__float_as_uint(x - hf) >> 16);  // x - hf exact (Sterbenz)
}

// ---------- prolog: row sums of squares of x ----------
__global__ __launch_bounds__(256) void row_sumsq(const float* __restrict__ X,
                                                 float* __restrict__ x2) {
    __shared__ float red[256];
    const int row = blockIdx.x;
    const float* p = X + (size_t)row * D_DIM;
    float s = 0.f;
    for (int c = threadIdx.x * 4; c < D_DIM; c += 256 * 4) {
        float4 v = *reinterpret_cast<const float4*>(p + c);
        s += v.x * v.x + v.y * v.y + v.z * v.z + v.w * v.w;
    }
    red[threadIdx.x] = s;
    __syncthreads();
    for (int off = 128; off > 0; off >>= 1) {
        if (threadIdx.x < off) red[threadIdx.x] += red[threadIdx.x + off];
        __syncthreads();
    }
    if (threadIdx.x == 0) x2[row] = red[0];
}

// ---------- prolog: column sums of squares of mu ----------
__global__ __launch_bounds__(256) void col_sumsq(const float* __restrict__ Mu,
                                                 float* __restrict__ mu2) {
    const int u = blockIdx.x * blockDim.x + threadIdx.x;
    if (u >= U_COLS) return;
    float s = 0.f;
    for (int d = 0; d < D_DIM; ++d) {
        float v = Mu[(size_t)d * U_COLS + u];
        s += v * v;
    }
    mu2[u] = s;
}

// ---------- prolog: split X into hi/lo bf16 planes (row-major) ----------
__global__ __launch_bounds__(256) void split_x(const float* __restrict__ X,
                                               unsigned short* __restrict__ Xhi,
                                               unsigned short* __restrict__ Xlo) {
    const size_t i = ((size_t)blockIdx.x * 256 + threadIdx.x) * 4;
    float4 v = *reinterpret_cast<const float4*>(X + i);
    float a[4] = {v.x, v.y, v.z, v.w};
    su4 h4, l4;
#pragma unroll
    for (int j = 0; j < 4; ++j) {
        unsigned short hh, ll;
        bsplit(a[j], hh, ll);
        h4[j] = hh; l4[j] = ll;
    }
    *reinterpret_cast<su4*>(Xhi + i) = h4;
    *reinterpret_cast<su4*>(Xlo + i) = l4;
}

// ---------- prolog: split + transpose mu -> MuT (U x D) hi/lo bf16 planes ----------
__global__ __launch_bounds__(256) void split_mu_t(const float* __restrict__ Mu,
                                                  unsigned short* __restrict__ MuThi,
                                                  unsigned short* __restrict__ MuTlo) {
    __shared__ unsigned short th[32][33];
    __shared__ unsigned short tl[32][33];
    const int tx = threadIdx.x & 31;
    const int ty = threadIdx.x >> 5;          // 0..7
    const int u0 = blockIdx.x * 32;
    const int d0 = blockIdx.y * 32;
#pragma unroll
    for (int i = 0; i < 4; ++i) {
        const int dl = ty + i * 8;
        float v = Mu[(size_t)(d0 + dl) * U_COLS + u0 + tx];
        unsigned short hh, ll;
        bsplit(v, hh, ll);
        th[dl][tx] = hh;
        tl[dl][tx] = ll;
    }
    __syncthreads();
#pragma unroll
    for (int i = 0; i < 4; ++i) {
        const int ul = ty + i * 8;
        MuThi[(size_t)(u0 + ul) * D_DIM + d0 + tx] = th[tx][ul];
        MuTlo[(size_t)(u0 + ul) * D_DIM + d0 + tx] = tl[tx][ul];
    }
}

// ---------- TDM 2D tile load: global (bf16) -> padded LDS ----------
__device__ __forceinline__ void tdm_load_2d(unsigned lds_byte_off,
                                            const void* gaddr,
                                            unsigned tensor_d0, unsigned tensor_d1,
                                            unsigned tile_d0, unsigned tile_d1,
                                            unsigned stride0_elems) {
#if HAVE_TDM
    unsigned long long ga = (unsigned long long)gaddr;
    v4u g0;
    g0[0] = 1u;                                            // count=1 valid descriptor
    g0[1] = lds_byte_off;                                  // lds_addr
    g0[2] = (unsigned)(ga & 0xffffffffu);                  // global_addr[31:0]
    g0[3] = (unsigned)((ga >> 32) & 0x01ffffffu) | (2u << 30);  // addr[56:32] | type=2
    v8i g1;
    const unsigned dw0 = (1u << 16)   // data_size = 1 -> 2-byte elements
                       | (1u << 20)   // pad_enable
                       | (3u << 22)   // pad_interval = 16 dwords (one 32-elem row)
                       | (3u << 25);  // pad_amount   = 4 dwords  (pitch 40 ushorts)
    g1[0] = (int)dw0;
    g1[1] = (int)((tensor_d0 & 0xffffu) << 16);            // tensor_dim0 lo16 @ bits[63:48]
    g1[2] = (int)(((tensor_d0 >> 16) & 0xffffu) | ((tensor_d1 & 0xffffu) << 16));
    g1[3] = (int)(((tensor_d1 >> 16) & 0xffffu) | ((tile_d0 & 0xffffu) << 16));
    g1[4] = (int)(tile_d1 & 0xffffu);                      // tile_dim1, tile_dim2=0
    g1[5] = (int)stride0_elems;                            // tensor_dim0_stride[31:0]
    g1[6] = 0;
    g1[7] = 0;
    v4i z4 = {0, 0, 0, 0};                                 // D# group 2 (unused, 2D)
    v8i z8 = {0, 0, 0, 0, 0, 0, 0, 0};                     // extra group (clang-23 form)
    __builtin_amdgcn_tensor_load_to_lds(g0, g1, z4, z4, z8, 0);
#else
    (void)lds_byte_off; (void)gaddr; (void)tensor_d0; (void)tensor_d1;
    (void)tile_d0; (void)tile_d1; (void)stride0_elems;
#endif
}

// ---------- shared compute: wave tile 64(M) x 32(N), bf16x3 WMMA ----------
__device__ __forceinline__ void wmma_slab(const unsigned short* s,
                                          int wm, int wn, int lrow, int lhalf,
                                          v8f acc[4][2]) {
    Frag ah[4], al[4], bh[2], bl[2];
#pragma unroll
    for (int t = 0; t < 4; ++t) {
        // A fragment (16x32 bf16): lanes 0-15 K{0..7,16..23}, lanes 16-31 K{8..15,24..31}
        int row = wm * 64 + t * 16 + lrow;
        ah[t].u[0] = *reinterpret_cast<const su8*>(s + A_HI + row * PITCH + lhalf * 8);
        ah[t].u[1] = *reinterpret_cast<const su8*>(s + A_HI + row * PITCH + 16 + lhalf * 8);
        al[t].u[0] = *reinterpret_cast<const su8*>(s + A_LO + row * PITCH + lhalf * 8);
        al[t].u[1] = *reinterpret_cast<const su8*>(s + A_LO + row * PITCH + 16 + lhalf * 8);
    }
#pragma unroll
    for (int t = 0; t < 2; ++t) {
        // B fragment (32x16 bf16): lanes 0-15 K{0..7,16..23}, lanes 16-31 K{8..15,24..31}, column = lrow
        int col = wn * 32 + t * 16 + lrow;
        const su8* pbh = reinterpret_cast<const su8*>(s + BT_HI + col * PITCH + lhalf * 8);
        const su8* pbl = reinterpret_cast<const su8*>(s + BT_LO + col * PITCH + lhalf * 8);
        bh[t].u[0] = pbh[0]; bh[t].u[1] = pbh[2];
        bl[t].u[0] = pbl[0]; bl[t].u[1] = pbl[2];
    }
#pragma unroll
    for (int mt = 0; mt < 4; ++mt)
#pragma unroll
        for (int nt = 0; nt < 2; ++nt) {
            v8f c = acc[mt][nt];
            c = __builtin_amdgcn_wmma_f32_16x16x32_bf16(false, ah[mt].v, false, bh[nt].v,
                                                        (short)0, c, false, false);
            c = __builtin_amdgcn_wmma_f32_16x16x32_bf16(false, ah[mt].v, false, bl[nt].v,
                                                        (short)0, c, false, false);
            c = __builtin_amdgcn_wmma_f32_16x16x32_bf16(false, al[mt].v, false, bh[nt].v,
                                                        (short)0, c, false, false);
            acc[mt][nt] = c;
        }
}

// ---------- shared epilogue ----------
__device__ __forceinline__ void rbf_epilogue(v8f acc[4][2], int m0, int n0,
                                             int wm, int wn, int lrow, int lhalf,
                                             float gamma,
                                             const float* __restrict__ x2,
                                             const float* __restrict__ mu2,
                                             float* __restrict__ out) {
#pragma unroll
    for (int mt = 0; mt < 4; ++mt)
#pragma unroll
        for (int nt = 0; nt < 2; ++nt) {
            const int col   = n0 + wn * 32 + nt * 16 + lrow;
            const int rbase = m0 + wm * 64 + mt * 16 + lhalf * 8;
            const float m2  = mu2[col];
            v8f c = acc[mt][nt];
#pragma unroll
            for (int v = 0; v < 8; ++v) {
                const int r = rbase + v;
                const float l2 = x2[r] - 2.0f * c[v] + m2;
                out[(size_t)r * U_COLS + col] = expf(-gamma * l2);
            }
        }
}

// ============ path A: TDM-fed GEMM, triple-buffered 2-deep pipeline ============
__global__ __launch_bounds__(256, 1) void rbf_gemm_tdm(
        const unsigned short* __restrict__ Xhi, const unsigned short* __restrict__ Xlo,
        const unsigned short* __restrict__ MuThi, const unsigned short* __restrict__ MuTlo,
        const float* __restrict__ gptr,
        const float* __restrict__ x2, const float* __restrict__ mu2,
        float* __restrict__ out) {
    __shared__ __align__(16) unsigned short lds[3 * STAGE];  // 120 KB, triple buffered

    const int tid   = threadIdx.x;
    const int lane  = tid & 31;
    const int wave  = tid >> 5;
    const int lrow  = lane & 15;
    const int lhalf = lane >> 4;
    const int wm    = wave & 1;    // 2 waves along M -> 64-row slices
    const int wn    = wave >> 1;   // 4 waves along N -> 32-col slices
    const int m0    = blockIdx.y * BM;
    const int n0    = blockIdx.x * BN;

    const float gamma = gptr[0];
    // low 32 bits of the generic address of an LDS object = byte offset in LDS
    const unsigned ldsbase = (unsigned)(unsigned long long)(const void*)lds;

    v8f acc[4][2] = {};

    auto stage_tdm = [&](int k0, int buf) {
        const unsigned sb = ldsbase + (unsigned)(buf * STAGE) * 2u;
        tdm_load_2d(sb + A_HI * 2u,  Xhi   + (size_t)m0 * D_DIM + k0,
                    D_DIM, B_ROWS, BK, BM, D_DIM);
        tdm_load_2d(sb + A_LO * 2u,  Xlo   + (size_t)m0 * D_DIM + k0,
                    D_DIM, B_ROWS, BK, BM, D_DIM);
        tdm_load_2d(sb + BT_HI * 2u, MuThi + (size_t)n0 * D_DIM + k0,
                    D_DIM, U_COLS, BK, BN, D_DIM);
        tdm_load_2d(sb + BT_LO * 2u, MuTlo + (size_t)n0 * D_DIM + k0,
                    D_DIM, U_COLS, BK, BN, D_DIM);
    };

    constexpr int KT = D_DIM / BK;  // 32 K-slabs
    // 2-deep prologue: slabs 0 and 1 in flight; wait until slab 0 lands (<=4 left)
    if (wave == 0) {
        stage_tdm(0, 0);
        stage_tdm(BK, 1);
#if HAVE_TDM && defined(__has_builtin) && __has_builtin(__builtin_amdgcn_s_wait_tensorcnt)
        __builtin_amdgcn_s_wait_tensorcnt(4);
#endif
    }
    __syncthreads();

    int cur = 0;
    for (int kb = 0; kb < KT; ++kb) {
        int nbuf = cur + 2; if (nbuf >= 3) nbuf -= 3;
        if (wave == 0 && kb + 2 < KT) stage_tdm((kb + 2) * BK, nbuf);
        wmma_slab(lds + cur * STAGE, wm, wn, lrow, lhalf, acc);
        if (wave == 0) {
#if HAVE_TDM && defined(__has_builtin) && __has_builtin(__builtin_amdgcn_s_wait_tensorcnt)
            if (kb + 2 < KT) __builtin_amdgcn_s_wait_tensorcnt(4);  // slab kb+1 done
            else             __builtin_amdgcn_s_wait_tensorcnt(0);  // drain tail
#endif
        }
        __syncthreads();
        cur = (cur == 2) ? 0 : cur + 1;
    }

    rbf_epilogue(acc, m0, n0, wm, wn, lrow, lhalf, gamma, x2, mu2, out);
}

// ============ path B (fallback): in-kernel split, fp32 operands ============
__global__ __launch_bounds__(256, 1) void rbf_gemm(const float* __restrict__ X,
                                                   const float* __restrict__ Mu,
                                                   const float* __restrict__ gptr,
                                                   const float* __restrict__ x2,
                                                   const float* __restrict__ mu2,
                                                   float* __restrict__ out) {
    __shared__ __align__(16) unsigned short lds[2 * STAGE];  // 80 KB, double buffered

    const int tid   = threadIdx.x;
    const int lane  = tid & 31;
    const int wave  = tid >> 5;
    const int lrow  = lane & 15;
    const int lhalf = lane >> 4;
    const int wm    = wave & 1;
    const int wn    = wave >> 1;
    const int m0    = blockIdx.y * BM;
    const int n0    = blockIdx.x * BN;

    const float gamma = gptr[0];

    v8f acc[4][2] = {};
    float4 aReg[4];
    float4 bReg[4];

    auto loadTile = [&](int k0) {
#pragma unroll
        for (int i = 0; i < 4; ++i) {            // A: 128x32 fp32 = 1024 float4
            int idx = tid + i * 256;
            int r = idx >> 3, c = (idx & 7) << 2;
            aReg[i] = *reinterpret_cast<const float4*>(X + (size_t)(m0 + r) * D_DIM + k0 + c);
        }
#pragma unroll
        for (int i = 0; i < 4; ++i) {            // B: 32x128 fp32 = 1024 float4
            int idx = tid + i * 256;
            int r = idx >> 5, c = (idx & 31) << 2;
            bReg[i] = *reinterpret_cast<const float4*>(Mu + (size_t)(k0 + r) * U_COLS + n0 + c);
        }
    };

    auto stageTile = [&](int buf) {
        unsigned short* s = lds + buf * STAGE;
#pragma unroll
        for (int i = 0; i < 4; ++i) {            // A -> LDS, row-major, split hi/lo
            int idx = tid + i * 256;
            int r = idx >> 3, c4 = idx & 7;
            float av[4] = {aReg[i].x, aReg[i].y, aReg[i].z, aReg[i].w};
            su4 h4, l4;
#pragma unroll
            for (int j = 0; j < 4; ++j) {
                unsigned short hh, ll;
                bsplit(av[j], hh, ll);
                h4[j] = hh; l4[j] = ll;
            }
            *reinterpret_cast<su4*>(s + A_HI + r * PITCH + c4 * 4) = h4;
            *reinterpret_cast<su4*>(s + A_LO + r * PITCH + c4 * 4) = l4;
        }
#pragma unroll
        for (int i = 0; i < 4; ++i) {            // B -> LDS transposed, split hi/lo
            int idx = tid + i * 256;
            int r = idx >> 5, c4 = idx & 31;
            float bv[4] = {bReg[i].x, bReg[i].y, bReg[i].z, bReg[i].w};
#pragma unroll
            for (int j = 0; j < 4; ++j) {
                int n = c4 * 4 + j;
                unsigned short hh, ll;
                bsplit(bv[j], hh, ll);
                s[BT_HI + n * PITCH + r] = hh;
                s[BT_LO + n * PITCH + r] = ll;
            }
        }
    };

    constexpr int KT = D_DIM / BK;
    loadTile(0);
    stageTile(0);
    __syncthreads();
    for (int kb = 0; kb < KT; ++kb) {
        if (kb + 1 < KT) loadTile((kb + 1) * BK);
        wmma_slab(lds + (kb & 1) * STAGE, wm, wn, lrow, lhalf, acc);
        __syncthreads();
        if (kb + 1 < KT) stageTile((kb + 1) & 1);
        __syncthreads();
    }

    rbf_epilogue(acc, m0, n0, wm, wn, lrow, lhalf, gamma, x2, mu2, out);
}

// ---------- host entry ----------
extern "C" void kernel_launch(void* const* d_in, const int* in_sizes, int n_in,
                              void* d_out, int out_size, void* d_ws, size_t ws_size,
                              hipStream_t stream) {
    const float* X  = (const float*)d_in[0];   // (8192, 1024)
    const float* Mu = (const float*)d_in[1];   // (1024, 2048)
    const float* g  = (const float*)d_in[2];   // scalar gamma
    float* out = (float*)d_out;                // (8192, 2048)

    float* x2  = (float*)d_ws;                 // 8192 floats
    float* mu2 = x2 + B_ROWS;                  // 2048 floats
    unsigned short* Xhi   = (unsigned short*)(mu2 + U_COLS);
    unsigned short* Xlo   = Xhi   + (size_t)B_ROWS * D_DIM;
    unsigned short* MuThi = Xlo   + (size_t)B_ROWS * D_DIM;
    unsigned short* MuTlo = MuThi + (size_t)U_COLS * D_DIM;

    const size_t need = (size_t)(B_ROWS + U_COLS) * sizeof(float)
                      + 2 * ((size_t)B_ROWS * D_DIM + (size_t)U_COLS * D_DIM)
                          * sizeof(unsigned short);

    row_sumsq<<<B_ROWS, 256, 0, stream>>>(X, x2);
    col_sumsq<<<U_COLS / 256, 256, 0, stream>>>(Mu, mu2);

    dim3 grid(U_COLS / BN, B_ROWS / BM);       // (16, 64)

    if (ws_size >= need) {
        split_x<<<(B_ROWS * D_DIM) / (256 * 4), 256, 0, stream>>>(X, Xhi, Xlo);
        dim3 tgrid(U_COLS / 32, D_DIM / 32);   // (64, 32)
        split_mu_t<<<tgrid, 256, 0, stream>>>(Mu, MuThi, MuTlo);
        rbf_gemm_tdm<<<grid, 256, 0, stream>>>(Xhi, Xlo, MuThi, MuTlo, g, x2, mu2, out);
    } else {
        rbf_gemm<<<grid, 256, 0, stream>>>(X, Mu, g, x2, mu2, out);
    }
}
